// LowRankOrthogonalMixer_37048387895884
// MI455X (gfx1250) — matrix operand order repaired and checked by reading, not compiled, not run on hardware
//
#include <hip/hip_runtime.h>

// ---------------------------------------------------------------------------
// LowRankOrthogonalMixer on MI455X (gfx1250, wave32, WMMA)
//
// Math: omega = U M U^T (U = [L|R|Ll|Rl], 512x64), Cayley transform via
// Woodbury:  transform = I + 2 U K U^T,  K = M' (I64 - G M')^{-1}, G = U^T U.
// out = x + (x U) V_b,  V_b = 2 K U^T  (precomputed per batch).
// Big GEMMs use V_WMMA_F32_16X16X32_BF16 with bf16 hi/lo operand splitting
// (3 WMMAs per logical fp32 product) for fp32-class accuracy at bf16 rate.
// ---------------------------------------------------------------------------

typedef __attribute__((ext_vector_type(16))) __bf16 v16bf;
typedef __attribute__((ext_vector_type(8)))  float  v8f;

#define F_DIM 512
#define N_DIM 4096
#define B_DIM 8

__device__ __forceinline__ float uget(const float* L, const float* R,
                                      const float* Ll, const float* Rl,
                                      int f, int j) {
  const float* m = (j < 16) ? L : (j < 32) ? R : (j < 48) ? Ll : Rl;
  return m[f * 16 + (j & 15)];
}

__device__ __forceinline__ void split_bf(float v, __bf16& h, __bf16& l) {
  h = (__bf16)v;
  l = (__bf16)(v - (float)h);
}

// --------------------------- G = U^T U (64x64) -----------------------------
__global__ void k_gram(const float* L, const float* R, const float* Ll,
                       const float* Rl, float* G) {
  int idx = blockIdx.x * 256 + threadIdx.x;   // 4096 entries
  int i = idx >> 6, j = idx & 63;
  float s = 0.f;
  for (int f = 0; f < F_DIM; ++f)
    s += uget(L, R, Ll, Rl, f, i) * uget(L, R, Ll, Rl, f, j);
  G[idx] = s;
}

// ------------- per-batch 64x64 Woodbury core solve: KT = K^T ---------------
// M'[i][j] closed form: 0.25*(Mg+Ml)[i][j] + (cs/24)*(MlGMg - MgGMl)[i][j]
__global__ __launch_bounds__(256) void k_solve(
    const float* G, const float* coeff, const float* gate,
    const float* coeffl, const float* gatel, const float* comm, float* KT) {
  __shared__ float Gs[64 * 64];
  __shared__ float A[64 * 64];
  __shared__ float Rs[64 * 64];   // holds M'^T, becomes K^T
  __shared__ float farr[64];
  __shared__ int piv;
  const int b = blockIdx.x, tid = threadIdx.x;
  for (int t = tid; t < 4096; t += 256) Gs[t] = G[t];
  const float g = gate[b], gl = gatel[b], cs = comm[b];
  const float* c  = coeff  + b * 16;
  const float* cl = coeffl + b * 16;
  __syncthreads();

  // Rs = M'^T  :  Rs[i][j] = M'[j][i]
  for (int t = tid; t < 4096; t += 256) {
    int mi = t & 63, mj = t >> 6;   // M' row mi, col mj ; stored transposed
    float base = 0.f;
    if (mi < 16 && mj == mi + 16)                    base += g * c[mi];
    else if (mi >= 16 && mi < 32 && mj == mi - 16)   base -= g * c[mj];
    else if (mi >= 32 && mi < 48 && mj == mi + 16)   base += gl * cl[mi - 32];
    else if (mi >= 48 && mj == mi - 16)              base -= gl * cl[mi - 48];
    float ta = 0.f, tb = 0.f;
    if (mi >= 32 && mj < 32) {                       // (Ml G Mg)[mi][mj]
      float sgn; int r, rho;
      if (mi < 48) { r = mi - 32; sgn =  1.f; rho = mi + 16; }
      else         { r = mi - 48; sgn = -1.f; rho = mi - 16; }
      if (mj < 16) ta = sgn * gl * cl[r] * Gs[rho * 64 + 16 + mj] * (-g * c[mj]);
      else { int jj = mj - 16;
             ta = sgn * gl * cl[r] * Gs[rho * 64 + jj] * (g * c[jj]); }
    }
    if (mi < 32 && mj >= 32) {                       // (Mg G Ml)[mi][mj]
      float ca; int rowg;
      if (mi < 16) { ca =  g * c[mi];      rowg = mi + 16; }
      else         { ca = -g * c[mi - 16]; rowg = mi - 16; }
      if (mj < 48) { int s = mj - 32; tb = ca * Gs[rowg * 64 + 48 + s] * (-gl * cl[s]); }
      else         { int s = mj - 48; tb = ca * Gs[rowg * 64 + 32 + s] * ( gl * cl[s]); }
    }
    Rs[(mj << 6) + mi] = 0.25f * base + (cs / 24.f) * (ta - tb);
  }
  __syncthreads();

  // A = I - Rs * Gs   (= (I - G M')^T since G symmetric)
  for (int t = tid; t < 4096; t += 256) {
    int i = t >> 6, j = t & 63;
    float acc = 0.f;
    for (int q = 0; q < 64; ++q) acc += Rs[i * 64 + q] * Gs[q * 64 + j];
    A[t] = ((i == j) ? 1.f : 0.f) - acc;
  }
  __syncthreads();

  // Gauss-Jordan with partial pivoting: A X = Rs  ->  Rs := K^T
  for (int k = 0; k < 64; ++k) {
    if (tid == 0) {
      int im = k; float mv = fabsf(A[k * 64 + k]);
      for (int i = k + 1; i < 64; ++i) {
        float v = fabsf(A[i * 64 + k]);
        if (v > mv) { mv = v; im = i; }
      }
      piv = im;
    }
    __syncthreads();
    int pr = piv;
    if (pr != k && tid < 128) {
      if (tid < 64) { float t0 = A[k*64+tid]; A[k*64+tid] = A[pr*64+tid]; A[pr*64+tid] = t0; }
      else { int jj = tid - 64; float t0 = Rs[k*64+jj]; Rs[k*64+jj] = Rs[pr*64+jj]; Rs[pr*64+jj] = t0; }
    }
    __syncthreads();
    float pivot = A[k * 64 + k];
    __syncthreads();
    float inv = 1.f / pivot;
    if (tid < 64) A[k * 64 + tid] *= inv;
    else if (tid < 128) Rs[k * 64 + (tid - 64)] *= inv;
    __syncthreads();
    if (tid < 64) farr[tid] = A[tid * 64 + k];
    __syncthreads();
    for (int t = tid; t < 8192; t += 256) {
      int i = t >> 7, jj = t & 127;
      if (i == k) continue;
      float f = farr[i];
      if (jj < 64) A[i * 64 + jj] -= f * A[k * 64 + jj];
      else Rs[i * 64 + (jj - 64)] -= f * Rs[k * 64 + (jj - 64)];
    }
    __syncthreads();
  }
  for (int t = tid; t < 4096; t += 256) KT[b * 4096 + t] = Rs[t];
}

// ---------------- pack U into stage-1 WMMA B-fragment order ----------------
// B layout (32x16 bf16): lane L, elem e -> K = e<8 ? (L>>4)*8+e : 16+(L>>4)*8+(e-8), n = L&15
__global__ void k_ufrag(const float* L, const float* R, const float* Ll,
                        const float* Rl, __bf16* UfH, __bf16* UfL) {
  int s = blockIdx.x * 256 + threadIdx.x;   // 32768 slots
  int e = s & 15, lane = (s >> 4) & 31, kk = (s >> 9) & 15, j = s >> 13;
  int f   = kk * 32 + (lane >> 4) * 8 + (e & 7) + (e >> 3) * 16;
  int col = j * 16 + (lane & 15);
  float v = uget(L, R, Ll, Rl, f, col);
  __bf16 h, l; split_bf(v, h, l);
  UfH[s] = h; UfL[s] = l;
}

// ---------------------- V[b] = 2 * K U^T  (64 x 512) -----------------------
__global__ void k_vmat(const float* KT, const float* L, const float* R,
                       const float* Ll, const float* Rl, float* V) {
  int idx = blockIdx.x * 256 + threadIdx.x;  // 4096 = b*512 + f
  int b = idx >> 9, f = idx & 511;
  float urow[64];
  #pragma unroll
  for (int j = 0; j < 64; ++j) urow[j] = uget(L, R, Ll, Rl, f, j);
  const float* Kb = KT + b * 4096;           // K^T
  for (int i = 0; i < 64; ++i) {
    float s = 0.f;
    for (int j = 0; j < 64; ++j) s += Kb[j * 64 + i] * urow[j];
    V[(b * 64 + i) * 512 + f] = 2.f * s;
  }
}

// ---------------- pack V into stage-2 WMMA B-fragment order ----------------
__global__ void k_vfrag(const float* V, __bf16* VfH, __bf16* VfL) {
  int s = blockIdx.x * 256 + threadIdx.x;    // 262144 slots
  int e = s & 15, lane = (s >> 4) & 31, kc = (s >> 9) & 1;
  int j = (s >> 10) & 31, b = s >> 15;
  int K = kc * 32 + (lane >> 4) * 8 + (e & 7) + (e >> 3) * 16;
  int f = j * 16 + (lane & 15);
  float v = V[(b * 64 + K) * 512 + f];
  __bf16 h, l; split_bf(v, h, l);
  VfH[s] = h; VfL[s] = l;
}

// --------------------------- main mixer kernel -----------------------------
// block = 8 waves, wave owns 16 rows.  P = x*U (K=512), out = x + P*V (K=64)
__global__ __launch_bounds__(256) void k_mixer(
    const float* __restrict__ x,
    const __bf16* __restrict__ UfH, const __bf16* __restrict__ UfL,
    const __bf16* __restrict__ VfH, const __bf16* __restrict__ VfL,
    float* __restrict__ out) {
  __shared__ float psh[8][16 * 68];          // padded: bank-conflict free
  const int tid = threadIdx.x, w = tid >> 5, lane = tid & 31;
  const int half = lane >> 4, lm = lane & 15;
  const int b = blockIdx.x >> 5, tile = blockIdx.x & 31;
  const int row0 = tile * 128 + w * 16;
  const v16bf* UH = (const v16bf*)UfH;
  const v16bf* UL = (const v16bf*)UfL;
  const v16bf* VH = (const v16bf*)VfH;
  const v16bf* VL = (const v16bf*)VfL;

  // ---------------- stage 1: P = x @ U, f32 accum, bf16x3 ----------------
  v8f acc[4] = {v8f{}, v8f{}, v8f{}, v8f{}};
  const size_t xrow = ((size_t)b * N_DIM + row0 + lm) * F_DIM;
  for (int kk = 0; kk < 16; ++kk) {
    // A-frag (16x32): lane holds row lm; e<8 -> K=half*8+e, e>=8 -> 16+half*8+(e-8)
    const float* px = x + xrow + kk * 32 + half * 8;
    v16bf ah, al;
    #pragma unroll
    for (int e = 0; e < 8; ++e) {
      __bf16 h, l;
      split_bf(px[e], h, l);       ah[e] = h;     al[e] = l;
      split_bf(px[16 + e], h, l);  ah[8 + e] = h; al[8 + e] = l;
    }
    #pragma unroll
    for (int j = 0; j < 4; ++j) {
      v16bf bh = UH[(j * 16 + kk) * 32 + lane];
      v16bf bl = UL[(j * 16 + kk) * 32 + lane];
      v8f a = acc[j];
      a = __builtin_amdgcn_wmma_f32_16x16x32_bf16(false, ah, false, bh, (short)0, a, false, false);
      a = __builtin_amdgcn_wmma_f32_16x16x32_bf16(false, al, false, bh, (short)0, a, false, false);
      a = __builtin_amdgcn_wmma_f32_16x16x32_bf16(false, ah, false, bl, (short)0, a, false, false);
      acc[j] = a;
    }
  }

  // ------- relayout P: C-layout (16x64) -> LDS -> A-layout fragments ------
  #pragma unroll
  for (int v = 0; v < 8; ++v) {
    int r = v + 8 * half;
    psh[w][r * 68 + 0  + lm] = acc[0][v];
    psh[w][r * 68 + 16 + lm] = acc[1][v];
    psh[w][r * 68 + 32 + lm] = acc[2][v];
    psh[w][r * 68 + 48 + lm] = acc[3][v];
  }
  __syncthreads();
  v16bf pah[2], pal[2];
  #pragma unroll
  for (int kc = 0; kc < 2; ++kc) {
    const float* q = &psh[w][lm * 68 + kc * 32 + half * 8];
    #pragma unroll
    for (int e = 0; e < 8; ++e) {
      __bf16 h, l;
      split_bf(q[e], h, l);       pah[kc][e] = h;     pal[kc][e] = l;
      split_bf(q[16 + e], h, l);  pah[kc][8 + e] = h; pal[kc][8 + e] = l;
    }
  }

  // ---------------- stage 2: out = x + P @ V_b (K=64) --------------------
  const size_t obase = ((size_t)b * N_DIM + row0) * F_DIM;
  for (int j2 = 0; j2 < 32; ++j2) {
    v8f c = {};
    #pragma unroll
    for (int kc = 0; kc < 2; ++kc) {
      size_t fi = ((size_t)((b * 32 + j2) * 2 + kc)) * 32 + lane;
      v16bf bh = VH[fi], bl = VL[fi];
      c = __builtin_amdgcn_wmma_f32_16x16x32_bf16(false, pah[kc], false, bh, (short)0, c, false, false);
      c = __builtin_amdgcn_wmma_f32_16x16x32_bf16(false, pal[kc], false, bh, (short)0, c, false, false);
      c = __builtin_amdgcn_wmma_f32_16x16x32_bf16(false, pah[kc], false, bl, (short)0, c, false, false);
    }
    int n = j2 * 16 + lm;
    #pragma unroll
    for (int v = 0; v < 8; ++v) {
      size_t idx = obase + (size_t)(v + 8 * half) * F_DIM + n;
      out[idx] = x[idx] + c[v];
    }
  }
}

// ---------------------------------------------------------------------------
extern "C" void kernel_launch(void* const* d_in, const int* in_sizes, int n_in,
                              void* d_out, int out_size, void* d_ws, size_t ws_size,
                              hipStream_t stream) {
  const float* x      = (const float*)d_in[0];
  const float* coeff  = (const float*)d_in[1];
  const float* gate   = (const float*)d_in[2];
  const float* coeffl = (const float*)d_in[3];
  const float* gatel  = (const float*)d_in[4];
  const float* comm   = (const float*)d_in[5];
  const float* L      = (const float*)d_in[6];
  const float* R      = (const float*)d_in[7];
  const float* Ll     = (const float*)d_in[8];
  const float* Rl     = (const float*)d_in[9];
  float* out = (float*)d_out;

  char* ws = (char*)d_ws;                       // ~2.3 MB used
  float*  G   = (float*)(ws);                   // 16 KB
  float*  KT  = (float*)(ws + 16384);           // 128 KB
  float*  V   = (float*)(ws + 147456);          // 1 MB
  __bf16* UfH = (__bf16*)(ws + 1196032);        // 64 KB
  __bf16* UfL = (__bf16*)(ws + 1261568);        // 64 KB
  __bf16* VfH = (__bf16*)(ws + 1327104);        // 512 KB
  __bf16* VfL = (__bf16*)(ws + 1851392);        // 512 KB

  k_gram <<<16,   256, 0, stream>>>(L, R, Ll, Rl, G);
  k_solve<<<B_DIM,256, 0, stream>>>(G, coeff, gate, coeffl, gatel, comm, KT);
  k_ufrag<<<128,  256, 0, stream>>>(L, R, Ll, Rl, UfH, UfL);
  k_vmat <<<16,   256, 0, stream>>>(KT, L, R, Ll, Rl, V);
  k_vfrag<<<1024, 256, 0, stream>>>(V, VfH, VfL);
  k_mixer<<<B_DIM * (N_DIM / 128), 256, 0, stream>>>(x, UfH, UfL, VfH, VfL, out);
}
